// NDT2Decoder_31963146617243
// MI455X (gfx1250) — matrix operand order repaired and checked by reading, not compiled, not run on hardware
//
#include <hip/hip_runtime.h>
#include <cstdint>
#include <cstddef>

// ---------------- problem constants ----------------
#define BB    16
#define LSEQ  1024
#define LQ    256
#define NCTX  8
#define BIN   256
#define DMODEL 1024
#define NH    16
#define DH    64
#define DFF   4096
#define DEPTH 6
#define LTOT  520          // NCTX + BIN + LQ
#define LPAD  544          // 32 * 17 (padded for 16/32 tiling)
#define MROWS (BB*LTOT)    // 8320 = 65*128
#define NEGB  (-1.0e9f)

typedef __attribute__((ext_vector_type(16))) __bf16       v16bf;
typedef __attribute__((ext_vector_type(8)))  float        v8f;
typedef __attribute__((ext_vector_type(8)))  unsigned int v8u;

// ---------------- helpers ----------------
// pack two fp32 -> two bf16 (round-half-up) in ONE v_perm_b32 + 2 adds
__device__ __forceinline__ unsigned int bfpack(float a, float b) {
  unsigned int au = __float_as_uint(a) + 0x8000u;
  unsigned int bu = __float_as_uint(b) + 0x8000u;
  return __builtin_amdgcn_perm(au, bu, 0x03020706u);  // lo16=bf(a), hi16=bf(b)
}
__device__ __forceinline__ unsigned short f2bfu(float f) {
  unsigned int u = __float_as_uint(f);
  return (unsigned short)((u + 0x8000u) >> 16);
}
__device__ __forceinline__ v16bf make16(uint4 a, uint4 b) {
  v8u w; w[0]=a.x; w[1]=a.y; w[2]=a.z; w[3]=a.w; w[4]=b.x; w[5]=b.y; w[6]=b.z; w[7]=b.w;
  return __builtin_bit_cast(v16bf, w);
}
__device__ __forceinline__ v8f v8zero() {
  v8f v = {0.f,0.f,0.f,0.f,0.f,0.f,0.f,0.f};
  return v;
}
__device__ __forceinline__ int imin(int a, int b) { return a < b ? a : b; }
__device__ __forceinline__ float gelu_exact(float x) {
  return 0.5f * x * (1.0f + erff(x * 0.70710678118654752f));
}

// =========================================================================
// fp32 -> bf16 bulk conversion (weights, once per call)
// =========================================================================
__global__ void cvt_bf16_kernel(const float* __restrict__ src,
                                unsigned short* __restrict__ dst, long np) {
  long i = (long)blockIdx.x * blockDim.x + threadIdx.x;   // pair index
  if (i < np) {
    float a = src[2*i], b = src[2*i+1];
    ((unsigned int*)dst)[i] = bfpack(a, b);
  }
}

// =========================================================================
// x construction
// =========================================================================
__global__ void build_ctx_kernel(const float* __restrict__ ctx, float* __restrict__ X) {
  int i = blockIdx.x, b = blockIdx.y;
  const float* src = ctx + ((size_t)b*NCTX + i) * DMODEL;
  float* dst = X + ((size_t)b*LTOT + i) * DMODEL;
  for (int d = threadIdx.x; d < DMODEL; d += 256) dst[d] = src[d];
}

// one block per (b, bin): deterministic binned mean pooling + time embedding
__global__ void build_pooled_kernel(const float* __restrict__ latents,
                                    const int* __restrict__ tidx,
                                    const unsigned char* __restrict__ in_mask,
                                    const float* __restrict__ time_w,
                                    float* __restrict__ X) {
  int s = blockIdx.x, b = blockIdx.y;
  __shared__ unsigned char flag[LSEQ];
  __shared__ int lst[LSEQ];
  __shared__ int cnt;
  for (int l = threadIdx.x; l < LSEQ; l += 256) {
    bool m = in_mask[(size_t)b*(NCTX+LSEQ) + NCTX + l] != 0;
    flag[l] = (m && tidx[(size_t)b*LSEQ + l] == s) ? 1 : 0;
  }
  __syncthreads();
  if (threadIdx.x == 0) {        // deterministic compaction (ordered)
    int c = 0;
    for (int l = 0; l < LSEQ; ++l) if (flag[l]) lst[c++] = l;
    cnt = c;
  }
  __syncthreads();
  int c = cnt;
  float inv = 1.0f / fmaxf((float)c, 1.0f);
  float* dst = X + ((size_t)b*LTOT + NCTX + s) * DMODEL;
  for (int d = threadIdx.x; d < DMODEL; d += 256) {
    float acc = 0.f;
    for (int i = 0; i < c; ++i)
      acc += latents[((size_t)b*LSEQ + lst[i]) * DMODEL + d];
    dst[d] = acc * inv + time_w[(size_t)s*DMODEL + d];
  }
}

__global__ void build_query_kernel(const int* __restrict__ qidx,
                                   const int* __restrict__ qtime,
                                   const float* __restrict__ qemb,
                                   const float* __restrict__ time_w,
                                   float* __restrict__ X) {
  int q = blockIdx.x, b = blockIdx.y;
  int qi = qidx[(size_t)b*LQ + q];
  int qt = qtime[(size_t)b*LQ + q];
  float* dst = X + ((size_t)b*LTOT + NCTX + BIN + q) * DMODEL;
  const float* e = qemb + (size_t)qi*DMODEL;
  const float* t = time_w + (size_t)qt*DMODEL;
  for (int d = threadIdx.x; d < DMODEL; d += 256) dst[d] = e[d] + t[d];
}

// per-position time value used for the causal/time attention bias
__global__ void build_tf_kernel(const int* __restrict__ qtime,
                                const unsigned char* __restrict__ qmask,
                                int* __restrict__ TF) {
  int b = blockIdx.x, p = threadIdx.x;   // 544 threads
  int v;
  if (p < NCTX)            v = -1;
  else if (p < NCTX+BIN)   v = p - NCTX;
  else if (p < LTOT) {
    int q = p - (NCTX+BIN);
    v = qmask[(size_t)b*LQ + q] ? qtime[(size_t)b*LQ + q] : 0x7FFFFFFF;
  } else                   v = 0x7FFFFFFF;
  TF[(size_t)b*LPAD + p] = v;
}

// =========================================================================
// LayerNorm, fp32 in -> bf16 out (consumed only as WMMA A operand)
// =========================================================================
__global__ __launch_bounds__(256)
void ln_kernel(const float* __restrict__ X, const float* __restrict__ sc,
               const float* __restrict__ bi, unsigned short* __restrict__ Y) {
  int row = blockIdx.x;
  const float* x = X + (size_t)row * DMODEL;
  unsigned int* y = (unsigned int*)(Y + (size_t)row * DMODEL);
  __shared__ float red[256];
  float s1 = 0.f, s2 = 0.f;
  for (int d = threadIdx.x; d < DMODEL; d += 256) { float v = x[d]; s1 += v; s2 += v*v; }
  red[threadIdx.x] = s1; __syncthreads();
  for (int o = 128; o > 0; o >>= 1) { if (threadIdx.x < o) red[threadIdx.x] += red[threadIdx.x+o]; __syncthreads(); }
  float mean = red[0] * (1.0f/DMODEL); __syncthreads();
  red[threadIdx.x] = s2; __syncthreads();
  for (int o = 128; o > 0; o >>= 1) { if (threadIdx.x < o) red[threadIdx.x] += red[threadIdx.x+o]; __syncthreads(); }
  float var = red[0] * (1.0f/DMODEL) - mean*mean;
  float rinv = rsqrtf(var + 1e-5f);
  for (int p = threadIdx.x; p < DMODEL/2; p += 256) {
    int d = 2*p;
    float a = (x[d]   - mean) * rinv * sc[d]   + bi[d];
    float b = (x[d+1] - mean) * rinv * sc[d+1] + bi[d+1];
    y[p] = bfpack(a, b);
  }
}

// =========================================================================
// bf16 WMMA GEMM:  Out[M,N] = epilogue( A[M,K] @ W[N,K]^T + bias[N] )
// A, W stored bf16; staging is pure b128 copies (no conversion in hot loop).
// mode 0: bias -> bf16 out | mode 1: bias + residual -> fp32 out
// mode 2: bias + exact GELU -> bf16 out
// Block tile 128x64, BK=32, 8 waves (4x2); each wave owns 32x32 -> 4 WMMA/step.
// =========================================================================
#define LDSS 40   // LDS row stride in bf16 elements (80B, 16B aligned rows)

__global__ __launch_bounds__(256)
void gemm_bf16_kernel(const unsigned short* __restrict__ A,
                      const unsigned short* __restrict__ W,
                      const float* __restrict__ bias,
                      const float* __restrict__ Res,
                      float* __restrict__ OutF,
                      unsigned short* __restrict__ OutH,
                      int M, int N, int K, int mode) {
  __shared__ unsigned short As[128 * LDSS];   // 10240 B
  __shared__ unsigned short Ws[64 * LDSS];    //  5120 B

  const int tid  = threadIdx.x;
  const int lane = tid & 31, wave = tid >> 5;
  const int wm = wave & 3, wn = wave >> 2;
  const int half = lane >> 4, m16 = lane & 15;
  const int bm = blockIdx.y, bn = blockIdx.x;

  const int arow = tid >> 1;                      // 0..127
  const int aseg = (tid & 1) * 16;                // 0 or 16 (shorts)
  const int wrow = tid >> 2;                      // 0..63
  const int wseg = (tid & 3) * 8;                 // 0,8,16,24 (shorts)

  v8f acc00 = v8zero(), acc01 = v8zero(), acc10 = v8zero(), acc11 = v8zero();

  const unsigned short* Ap = A + (size_t)(bm*128 + arow) * K + aseg;
  const unsigned short* Wp = W + (size_t)(bn*64  + wrow) * K + wseg;

  for (int k0 = 0; k0 < K; k0 += 32) {
    __syncthreads();
    { // stage A tile: 16 shorts/thread, raw copies
      uint4 p0 = *(const uint4*)(Ap + k0);
      uint4 p1 = *(const uint4*)(Ap + k0 + 8);
      *(uint4*)&As[arow*LDSS + aseg]     = p0;
      *(uint4*)&As[arow*LDSS + aseg + 8] = p1;
    }
    { // stage W tile: 8 shorts/thread
      *(uint4*)&Ws[wrow*LDSS + wseg] = *(const uint4*)(Wp + k0);
    }
    if (k0 + 32 < K) {                 // pull next K tile toward the WGP
      __builtin_prefetch(Ap + k0 + 32, 0, 1);
      __builtin_prefetch(Wp + k0 + 32, 0, 1);
    }
    __syncthreads();

    const unsigned short* ap0 = &As[(wm*32 +      m16) * LDSS];
    const unsigned short* ap1 = &As[(wm*32 + 16 + m16) * LDSS];
    v16bf a0 = make16(*(const uint4*)(ap0 + half*16), *(const uint4*)(ap0 + half*16 + 8));
    v16bf a1 = make16(*(const uint4*)(ap1 + half*16), *(const uint4*)(ap1 + half*16 + 8));
    const unsigned short* bp0 = &Ws[(wn*32 +      m16) * LDSS + half*16];
    const unsigned short* bp1 = &Ws[(wn*32 + 16 + m16) * LDSS + half*16];
    v16bf b0 = make16(*(const uint4*)bp0, *(const uint4*)(bp0 + 8));
    v16bf b1 = make16(*(const uint4*)bp1, *(const uint4*)(bp1 + 8));

    acc00 = __builtin_amdgcn_wmma_f32_16x16x32_bf16(false, a0, false, b0, (short)0, acc00, false, false);
    acc01 = __builtin_amdgcn_wmma_f32_16x16x32_bf16(false, a0, false, b1, (short)0, acc01, false, false);
    acc10 = __builtin_amdgcn_wmma_f32_16x16x32_bf16(false, a1, false, b0, (short)0, acc10, false, false);
    acc11 = __builtin_amdgcn_wmma_f32_16x16x32_bf16(false, a1, false, b1, (short)0, acc11, false, false);
  }

  // epilogue (C layout: vgpr r -> row r + 8*half, col = lane&15)
  const int row0 = bm*128 + wm*32 + half*8;
  const int col0 = bn*64  + wn*32 + m16;
  #pragma unroll
  for (int q = 0; q < 4; ++q) {
    v8f acc = (q==0) ? acc00 : (q==1) ? acc01 : (q==2) ? acc10 : acc11;
    int col = col0 + ((q & 1) ? 16 : 0);
    int rw  = row0 + ((q & 2) ? 16 : 0);
    float bval = bias[col];
    #pragma unroll
    for (int r = 0; r < 8; ++r) {
      float v = acc[r] + bval;
      size_t o = (size_t)(rw + r) * N + col;
      if (mode == 1)      OutF[o] = v + Res[o];
      else if (mode == 2) OutH[o] = f2bfu(gelu_exact(v));
      else                OutH[o] = f2bfu(v);
    }
  }
}

// =========================================================================
// Fused flash attention: one wave handles 16 query rows of one (b,h).
// QKV stored bf16: [b][pos][{q,k,v} x 1024]; output bf16 (feeds out-proj GEMM).
// =========================================================================
__global__ __launch_bounds__(32)
void attn_kernel(const unsigned short* __restrict__ QKV, const int* __restrict__ TF,
                 unsigned short* __restrict__ Outp) {
  const int mt = blockIdx.x;          // 0..33 row tiles (LPAD/16)
  const int h  = blockIdx.y;
  const int b  = blockIdx.z;
  const int lane = threadIdx.x;
  const int half = lane >> 4, m16 = lane & 15;

  __shared__ unsigned short Plds[16 * LDSS];   // P tile for C->A layout transpose

  const size_t base = (size_t)b * LTOT * (3*DMODEL);
  const int* tf = TF + (size_t)b * LPAD;

  // ---- Q fragments (A-matrix, 16 rows x 32 K), two dh-halves: raw bf16 loads
  int qrow = imin(mt*16 + m16, LTOT-1);
  const unsigned short* qp = QKV + base + (size_t)qrow * (3*DMODEL) + h*DH;
  v16bf qa[2];
  #pragma unroll
  for (int ks = 0; ks < 2; ++ks) {
    const unsigned short* p0 = qp + ks*32 + half*16;
    const unsigned short* p1 = qp + ks*32 + half*16 + 8;
    qa[ks] = make16(*(const uint4*)p0, *(const uint4*)p1);
  }

  int trow[8];
  #pragma unroll
  for (int r = 0; r < 8; ++r) trow[r] = tf[mt*16 + r + half*8];

  float mrow[8], lrw[8];
  v8f o[4];
  #pragma unroll
  for (int r = 0; r < 8; ++r) { mrow[r] = -3.0e38f; lrw[r] = 0.f; }
  #pragma unroll
  for (int f = 0; f < 4; ++f) o[f] = v8zero();

  for (int kc = 0; kc < LPAD/32; ++kc) {
    const int j0 = kc * 32;
    float s0[8], s1[8];

    #pragma unroll
    for (int cn = 0; cn < 2; ++cn) {
      v8f accS = v8zero();
      int key  = j0 + cn*16 + m16;
      int keyc = imin(key, LTOT-1);
      const unsigned short* kp = QKV + base + (size_t)keyc * (3*DMODEL) + DMODEL + h*DH;
      #pragma unroll
      for (int ks = 0; ks < 2; ++ks) {
        const unsigned short* p = kp + ks*32 + half*16;   // 16 contiguous bf16
        v16bf kb = make16(*(const uint4*)p, *(const uint4*)(p+8));
        accS = __builtin_amdgcn_wmma_f32_16x16x32_bf16(false, qa[ks], false, kb, (short)0, accS, false, false);
      }
      int tcol = tf[key];
      #pragma unroll
      for (int r = 0; r < 8; ++r) {
        int row = mt*16 + r + half*8;
        bool blk = (key >= NCTX) && ((row < NCTX) || (trow[r] < tcol));
        float v = accS[r] * 0.125f + (blk ? NEGB : 0.f);
        if (cn) s1[r] = v; else s0[r] = v;
      }
    }

    // online softmax update + stash P (bf16) into LDS
    #pragma unroll
    for (int r = 0; r < 8; ++r) {
      float v = fmaxf(s0[r], s1[r]);
      #pragma unroll
      for (int off = 1; off < 16; off <<= 1) v = fmaxf(v, __shfl_xor(v, off, 32));
      float mn = fmaxf(mrow[r], v);
      float p0 = __expf(s0[r] - mn);
      float p1 = __expf(s1[r] - mn);
      float rs = p0 + p1;
      #pragma unroll
      for (int off = 1; off < 16; off <<= 1) rs += __shfl_xor(rs, off, 32);
      float scl = __expf(mrow[r] - mn);
      lrw[r] = lrw[r] * scl + rs;
      mrow[r] = mn;
      #pragma unroll
      for (int f = 0; f < 4; ++f) o[f][r] *= scl;
      int prow = r + half*8;
      Plds[prow*LDSS + m16]      = f2bfu(p0);
      Plds[prow*LDSS + 16 + m16] = f2bfu(p1);
    }
    __syncthreads();

    // P as A-matrix fragment (16 rows x 32 keys)
    const unsigned short* pp = &Plds[m16 * LDSS];
    v16bf pa = make16(*(const uint4*)(pp + half*16),
                      *(const uint4*)(pp + half*16 + 8));

    // O += P @ V  (4 dh-column fragments); V gathers are raw bf16 u16 loads
    #pragma unroll
    for (int f = 0; f < 4; ++f) {
      v8u w;
      const unsigned short* vp = QKV + base + 2*DMODEL + h*DH + f*16 + m16;
      #pragma unroll
      for (int e2 = 0; e2 < 8; ++e2) {
        int k1 = imin(j0 + half*16 + 2*e2,     LTOT-1);
        int k2 = imin(j0 + half*16 + 2*e2 + 1, LTOT-1);
        unsigned int va = vp[(size_t)k1 * (3*DMODEL)];
        unsigned int vb = vp[(size_t)k2 * (3*DMODEL)];
        w[e2] = va | (vb << 16);
      }
      v16bf vb16 = __builtin_bit_cast(v16bf, w);
      o[f] = __builtin_amdgcn_wmma_f32_16x16x32_bf16(false, pa, false, vb16, (short)0, o[f], false, false);
    }
    __syncthreads();
  }

  // write O / l  (bf16, consumed by out-proj GEMM)
  #pragma unroll
  for (int r = 0; r < 8; ++r) {
    int row = mt*16 + r + half*8;
    if (row < LTOT) {
      float inv = 1.0f / lrw[r];
      unsigned short* op = Outp + ((size_t)b*LTOT + row) * DMODEL + h*DH + m16;
      op[0]  = f2bfu(o[0][r] * inv);
      op[16] = f2bfu(o[1][r] * inv);
      op[32] = f2bfu(o[2][r] * inv);
      op[48] = f2bfu(o[3][r] * inv);
    }
  }
}

// =========================================================================
__global__ void copy_out_kernel(const float* __restrict__ X, float* __restrict__ out) {
  size_t i = (size_t)blockIdx.x * blockDim.x + threadIdx.x;
  const size_t total = (size_t)BB * (LTOT-NCTX) * DMODEL;   // 16*512*1024
  if (i < total) {
    int d = (int)(i & (DMODEL-1));
    size_t row = i >> 10;
    int b = (int)(row >> 9);
    int rr = (int)(row & 511);
    out[i] = X[((size_t)b*LTOT + NCTX + rr) * DMODEL + d];
  }
}

// =========================================================================
extern "C" void kernel_launch(void* const* d_in, const int* in_sizes, int n_in,
                              void* d_out, int out_size, void* d_ws, size_t ws_size,
                              hipStream_t stream) {
  const float*         latents   = (const float*)d_in[0];
  const int*           in_tidx   = (const int*)d_in[1];
  const unsigned char* in_mask   = (const unsigned char*)d_in[3];
  const int*           qidx      = (const int*)d_in[4];
  const int*           qtime     = (const int*)d_in[5];
  const unsigned char* qmask     = (const unsigned char*)d_in[7];
  const float*         ctx_emb   = (const float*)d_in[8];
  const float*         qemb_w    = (const float*)d_in[9];
  const float*         time_w    = (const float*)d_in[10];
  const float*         qkv_w     = (const float*)d_in[11];
  const float*         qkv_b     = (const float*)d_in[12];
  const float*         out_w     = (const float*)d_in[13];
  const float*         out_b     = (const float*)d_in[14];
  const float*         ln1_s     = (const float*)d_in[15];
  const float*         ln1_b     = (const float*)d_in[16];
  const float*         ln2_s     = (const float*)d_in[17];
  const float*         ln2_b     = (const float*)d_in[18];
  const float*         w1        = (const float*)d_in[19];
  const float*         b1        = (const float*)d_in[20];
  const float*         w2        = (const float*)d_in[21];
  const float*         b2        = (const float*)d_in[22];

  // ---- workspace layout (~340MB) ----
  float* X = (float*)d_ws;                                        // (B,520,1024) fp32
  unsigned short* Ybf  = (unsigned short*)(X + (size_t)BB*LTOT*DMODEL);
  unsigned short* QKVb = Ybf  + (size_t)BB*LTOT*DMODEL;           // (B,520,3072) bf16
  unsigned short* ATTb = QKVb + (size_t)BB*LTOT*3*DMODEL;         // (B,520,1024) bf16
  unsigned short* FFb  = ATTb + (size_t)BB*LTOT*DMODEL;           // (B,520,4096) bf16
  unsigned short* Wq   = FFb  + (size_t)BB*LTOT*DFF;              // bf16 weights
  unsigned short* Wo   = Wq   + (size_t)DEPTH*3*DMODEL*DMODEL;
  unsigned short* W1   = Wo   + (size_t)DEPTH*DMODEL*DMODEL;
  unsigned short* W2   = W1   + (size_t)DEPTH*DFF*DMODEL;
  int*            TF   = (int*)(W2 + (size_t)DEPTH*DMODEL*DFF);

  // ---- one-time per-call weight conversion to bf16 ----
  {
    long nq = (long)DEPTH*3*DMODEL*DMODEL/2, no = (long)DEPTH*DMODEL*DMODEL/2;
    long n1 = (long)DEPTH*DFF*DMODEL/2,      n2 = (long)DEPTH*DMODEL*DFF/2;
    cvt_bf16_kernel<<<(unsigned)((nq+255)/256), 256, 0, stream>>>(qkv_w, Wq, nq);
    cvt_bf16_kernel<<<(unsigned)((no+255)/256), 256, 0, stream>>>(out_w, Wo, no);
    cvt_bf16_kernel<<<(unsigned)((n1+255)/256), 256, 0, stream>>>(w1,    W1, n1);
    cvt_bf16_kernel<<<(unsigned)((n2+255)/256), 256, 0, stream>>>(w2,    W2, n2);
  }

  build_ctx_kernel   <<<dim3(NCTX, BB), 256, 0, stream>>>(ctx_emb, X);
  build_pooled_kernel<<<dim3(BIN,  BB), 256, 0, stream>>>(latents, in_tidx, in_mask, time_w, X);
  build_query_kernel <<<dim3(LQ,   BB), 256, 0, stream>>>(qidx, qtime, qemb_w, time_w, X);
  build_tf_kernel    <<<BB, LPAD, 0, stream>>>(qtime, qmask, TF);

  for (int l = 0; l < DEPTH; ++l) {
    ln_kernel<<<MROWS, 256, 0, stream>>>(X, ln1_s + (size_t)l*DMODEL, ln1_b + (size_t)l*DMODEL, Ybf);
    gemm_bf16_kernel<<<dim3(3*DMODEL/64, MROWS/128), 256, 0, stream>>>(
        Ybf, Wq + (size_t)l*3*DMODEL*DMODEL, qkv_b + (size_t)l*3*DMODEL,
        nullptr, nullptr, QKVb, MROWS, 3*DMODEL, DMODEL, 0);
    attn_kernel<<<dim3(LPAD/16, NH, BB), 32, 0, stream>>>(QKVb, TF, ATTb);
    gemm_bf16_kernel<<<dim3(DMODEL/64, MROWS/128), 256, 0, stream>>>(
        ATTb, Wo + (size_t)l*DMODEL*DMODEL, out_b + (size_t)l*DMODEL,
        X, X, nullptr, MROWS, DMODEL, DMODEL, 1);
    ln_kernel<<<MROWS, 256, 0, stream>>>(X, ln2_s + (size_t)l*DMODEL, ln2_b + (size_t)l*DMODEL, Ybf);
    gemm_bf16_kernel<<<dim3(DFF/64, MROWS/128), 256, 0, stream>>>(
        Ybf, W1 + (size_t)l*DFF*DMODEL, b1 + (size_t)l*DFF,
        nullptr, nullptr, FFb, MROWS, DFF, DMODEL, 2);
    gemm_bf16_kernel<<<dim3(DMODEL/64, MROWS/128), 256, 0, stream>>>(
        FFb, W2 + (size_t)l*DMODEL*DFF, b2 + (size_t)l*DMODEL,
        X, X, nullptr, MROWS, DMODEL, DFF, 1);
  }

  const size_t total = (size_t)BB * (LTOT-NCTX) * DMODEL;
  copy_out_kernel<<<(unsigned)((total + 255) / 256), 256, 0, stream>>>(X, (float*)d_out);
}
